// OneD_RNN_wavefxn_12025908429426
// MI455X (gfx1250) — compile-verified, matrix operand order checked by reading, not run on hardware
//
#include <hip/hip_runtime.h>

typedef __attribute__((ext_vector_type(16))) _Float16 v16h;
typedef __attribute__((ext_vector_type(8)))  _Float16 v8h;
typedef __attribute__((ext_vector_type(8)))  float    v8f;

#define NH       256
#define NG       768      // 3*NH
#define NSTEPS   256
#define ROWS     16       // batch rows per workgroup (one WMMA M-tile)
#define NSAMPLES 2048
#define NTILES   48       // NG / 16
#define KTILES   8        // NH / 32

__device__ __forceinline__ unsigned int hash32(unsigned int a) {
  a ^= a >> 16; a *= 0x7feb352du;
  a ^= a >> 15; a *= 0x846ca68bu;
  a ^= a >> 16;
  return a;
}

__device__ __forceinline__ float fast_sigmoid(float x) {
  return 1.f / (1.f + __expf(-x));
}
__device__ __forceinline__ float fast_tanh(float x) {
  // tanh(x) = 1 - 2/(exp(2x)+1)   (single v_exp_f32 + v_rcp)
  return 1.f - 2.f / (__expf(2.f * x) + 1.f);
}

// Pack Wr (fp32 [NH][NG] row-major) into f16 WMMA B-fragments.
// Fragment f = ntile*KTILES + kt holds the 32x16 (KxN) tile in the per-lane
// layout of V_WMMA_F32_16X16X32_F16 (ISA 7.12.2): lane -> N = lane&15,
// half hi -> K = hi + (hi>=8 ? 8:0) + (lane>=16 ? 8:0).
__global__ void pack_wr_kernel(const float* __restrict__ Wr,
                               _Float16* __restrict__ wsB) {
  const int f     = blockIdx.x;            // 0 .. NTILES*KTILES-1
  const int ntile = f >> 3;
  const int kt    = f & 7;
  const int lane  = threadIdx.x;           // 0..31
  const int n     = ntile * 16 + (lane & 15);
  const int kadd  = (lane >= 16) ? 8 : 0;
  _Float16* dst = wsB + (size_t)f * 512 + lane * 16;
#pragma unroll
  for (int hi = 0; hi < 16; ++hi) {
    const int k = kt * 32 + hi + ((hi >= 8) ? 8 : 0) + kadd;
    dst[hi] = (_Float16)Wr[k * NG + n];
  }
}

__global__ __launch_bounds__(256, 1) void gru_sampler_kernel(
    const float* __restrict__ Wk,    // [2][NG]
    const float* __restrict__ bias,  // [2][NG] (b0 row then b1 row)
    const float* __restrict__ Wd,    // [NH][2]
    const float* __restrict__ bd,    // [2]
    const _Float16* __restrict__ wsB,
    float* __restrict__ out) {
  __shared__ _Float16 h_lds[ROWS][264];   // hidden state, f16, 16B-aligned rows
  __shared__ _Float16 gh_lds[ROWS][776];  // gate pre-activations (h @ Wr)
  // comb[idx][col]: x-side gate constants, b1 folded in for z/r columns.
  //   idx 0: x = 0 (step 0);  idx 1: x = onehot(0);  idx 2: x = onehot(1)
  __shared__ float comb_s[3 * NG];
  __shared__ float b1n_s[NH];             // recurrent bias of the n-gate
  __shared__ float wd_s[NH * 2];
  __shared__ float bd_s[2];
  __shared__ int   s_prev[ROWS];

  const int tid  = threadIdx.x;
  const int wave = tid >> 5;
  const int lane = tid & 31;

  for (int i = tid; i < ROWS * 264; i += 256) (&h_lds[0][0])[i] = (_Float16)0.f;
  for (int i = tid; i < NG; i += 256) {
    const float b0v = bias[i];
    const float b1v = bias[NG + i];
    const float base = b0v + ((i < 2 * NH) ? b1v : 0.f);  // fold b1 into z/r
    comb_s[0 * NG + i] = base;
    comb_s[1 * NG + i] = base + Wk[i];
    comb_s[2 * NG + i] = base + Wk[NG + i];
    if (i >= 2 * NH) b1n_s[i - 2 * NH] = b1v;
  }
  for (int i = tid; i < NH * 2; i += 256) wd_s[i] = Wd[i];
  if (tid < 2)    bd_s[tid]   = bd[tid];
  if (tid < ROWS) s_prev[tid] = 0;
  __syncthreads();

  const int row = tid >> 4;   // 0..15
  const int c0  = tid & 15;   // column phase within row
  const int sg  = blockIdx.x * ROWS + row;  // global sample index
  float logP = 0.f;
  int*   samp    = (int*)out;                       // samples [NSAMPLES][NSTEPS] i32
  float* logPout = out + (size_t)NSAMPLES * NSTEPS; // logP [NSAMPLES] f32

  // WMMA per-lane addressing constants
  const int am    = lane & 15;              // A: M = lane&15
  const int kadd  = (lane >= 16) ? 8 : 0;   // A/B K sub-offset
  const int nn    = lane & 15;              // C/D: N = lane&15
  const int mbase = (lane >= 16) ? 8 : 0;   // C/D: M = vgpr + mbase

  for (int step = 0; step < NSTEPS; ++step) {
    // -------- recurrent GEMM: gh = h @ Wr via v_wmma_f32_16x16x32_f16 --------
    v16h A[KTILES];
#pragma unroll
    for (int kt = 0; kt < KTILES; ++kt) {
      union { v16h v; v8h h[2]; } u;
      u.h[0] = *(const v8h*)&h_lds[am][kt * 32 + kadd];        // K = {0..7}+kadd
      u.h[1] = *(const v8h*)&h_lds[am][kt * 32 + 16 + kadd];   // K = {16..23}+kadd
      A[kt] = u.v;
    }
#pragma unroll 1
    for (int nt = 0; nt < 6; ++nt) {
      const int ntile = wave * 6 + nt;
      const _Float16* bp = wsB + (size_t)ntile * (KTILES * 512) + lane * 16;
      // locality 3 -> WGP-scope prefetch: pulls next tile's lines into WGP$
      __builtin_prefetch(bp + KTILES * 512, 0, 3);
      v8f c = {};
#pragma unroll
      for (int kt = 0; kt < KTILES; ++kt) {
        const v16h B = *(const v16h*)(bp + kt * 512);
        c = __builtin_amdgcn_wmma_f32_16x16x32_f16(
                false, A[kt], false, B, (short)0, c, false, false);
      }
      const int ncol = ntile * 16 + nn;
#pragma unroll
      for (int v = 0; v < 8; ++v)
        gh_lds[mbase + v][ncol] = (_Float16)c[v];
    }
    __syncthreads();

    // -------- gates (z, r, h) + hidden update --------
    const int idx = (step == 0) ? 0 : (1 + s_prev[row]);
    const float* cmb = &comb_s[idx * NG];
    float hbuf[16];
#pragma unroll
    for (int i = 0; i < 16; ++i) {
      const int j = c0 + i * 16;
      const float z  = fast_sigmoid((float)gh_lds[row][j]      + cmb[j]);
      const float r  = fast_sigmoid((float)gh_lds[row][NH + j] + cmb[NH + j]);
      const float hn = (float)gh_lds[row][2 * NH + j] + b1n_s[j];
      const float hh = fast_tanh(cmb[2 * NH + j] + r * hn);
      const float ho = (float)h_lds[row][j];
      const float hv = z * ho + (1.f - z) * hh;
      h_lds[row][j] = (_Float16)hv;
      hbuf[i] = hv;
    }

    // -------- dense head (NH -> 2), softmax, categorical sample --------
    float p0 = 0.f, p1 = 0.f;
#pragma unroll
    for (int i = 0; i < 16; ++i) {
      const int j = c0 + i * 16;
      p0 += hbuf[i] * wd_s[j * 2 + 0];
      p1 += hbuf[i] * wd_s[j * 2 + 1];
    }
#pragma unroll
    for (int m = 8; m >= 1; m >>= 1) {
      p0 += __shfl_xor(p0, m, 32);
      p1 += __shfl_xor(p1, m, 32);
    }
    if (c0 == 0) {
      const float l0 = p0 + bd_s[0], l1 = p1 + bd_s[1];
      const float mx = fmaxf(l0, l1);
      const float e0 = __expf(l0 - mx), e1 = __expf(l1 - mx);
      const float inv = 1.f / (e0 + e1);
      const float q0 = e0 * inv, q1 = e1 * inv;
      const unsigned int hs =
          hash32((unsigned int)sg * 2654435761u ^
                 ((unsigned int)step * 0x85ebca6bu + 0x9e3779b9u));
      const float u = (float)(hs >> 8) * (1.0f / 16777216.0f);
      const int s = (u < q0) ? 0 : 1;
      logP += logf(1e-10f + (s ? q1 : q0));
      samp[(size_t)sg * NSTEPS + step] = s;
      s_prev[row] = s;
    }
    __syncthreads();
  }
  if (c0 == 0) logPout[sg] = logP;
}

extern "C" void kernel_launch(void* const* d_in, const int* in_sizes, int n_in,
                              void* d_out, int out_size, void* d_ws, size_t ws_size,
                              hipStream_t stream) {
  (void)in_sizes; (void)n_in; (void)out_size; (void)ws_size;
  const float* Wk = (const float*)d_in[0];  // gru_kernel      [2, 768]
  const float* Wr = (const float*)d_in[1];  // gru_rec_kernel  [256, 768]
  const float* b  = (const float*)d_in[2];  // gru_bias        [2, 768]
  const float* Wd = (const float*)d_in[3];  // dense_kernel    [256, 2]
  const float* bd = (const float*)d_in[4];  // dense_bias      [2]
  _Float16* wsB = (_Float16*)d_ws;          // 384 KB packed Wr fragments

  pack_wr_kernel<<<NTILES * KTILES, 32, 0, stream>>>(Wr, wsB);
  gru_sampler_kernel<<<NSAMPLES / ROWS, 256, 0, stream>>>(
      Wk, b, Wd, bd, wsB, (float*)d_out);
}
